// Head_20332375179864
// MI455X (gfx1250) — matrix operand order repaired and checked by reading, not compiled, not run on hardware
//
#include <hip/hip_runtime.h>
#include <hip/hip_bf16.h>

#define B_  8
#define T_  2048
#define D_  768
#define HS_ 64

typedef __attribute__((ext_vector_type(16))) __bf16 v16bf;
typedef __attribute__((ext_vector_type(8)))  __bf16 v8bf;
typedef __attribute__((ext_vector_type(8)))  float  v8f;

union V16U { v16bf v; v8bf h[2]; };

// Workspace layout (bytes):
//   Wt  : [3][64][768] bf16 (transposed weights)          @ 0
//   Q   : [B][T][64]   bf16 (pre-scaled by 1/8)           @ WS_Q
//   K   : [B][T][64]   bf16                               @ WS_K
//   Vt  : [B][64][T]   bf16 (transposed values)           @ WS_V
static constexpr size_t WS_WT = 0;
static constexpr size_t WS_Q  = (size_t)3 * HS_ * D_ * 2;              // 294912
static constexpr size_t WS_K  = WS_Q + (size_t)B_ * T_ * HS_ * 2;
static constexpr size_t WS_V  = WS_K + (size_t)B_ * T_ * HS_ * 2;

// ---------------------------------------------------------------------------
// Kernel 0: transpose + bf16-convert the three weight matrices.
// Wt[m][h][d] = W_m[d][h]
// ---------------------------------------------------------------------------
__global__ void repack_w(const float* __restrict__ Wq,
                         const float* __restrict__ Wk,
                         const float* __restrict__ Wv,
                         __bf16* __restrict__ Wt) {
  int idx = blockIdx.x * blockDim.x + threadIdx.x;
  if (idx >= 3 * HS_ * D_) return;
  int m = idx / (HS_ * D_);
  int r = idx % (HS_ * D_);
  int h = r / D_;
  int d = r % D_;
  const float* W = (m == 0) ? Wq : (m == 1) ? Wk : Wv;
  Wt[idx] = (__bf16)W[d * HS_ + h];
}

// ---------------------------------------------------------------------------
// Kernel 1: QKV projection via v_wmma_f32_16x16x32_bf16.
// One wave computes a 16-token x 64-col tile of one of {Q,K,V}.
// ---------------------------------------------------------------------------
__global__ void __launch_bounds__(256) qkv_proj(const float*  __restrict__ x,
                                                const __bf16* __restrict__ Wt,
                                                __bf16* __restrict__ qb,
                                                __bf16* __restrict__ kb,
                                                __bf16* __restrict__ vT) {
  const int lane = threadIdx.x & 31;
  const int half = lane >> 4;
  const int ln   = lane & 15;
  // wave-uniform scalars -> SGPRs so Q/K/V dispatch is a scalar branch
  const int wid  = __builtin_amdgcn_readfirstlane(blockIdx.x * 8 + (threadIdx.x >> 5));
  const int m    = wid % 3;                        // 0=Q 1=K 2=V
  const int tile = wid / 3;                        // 0..1023
  const int b    = tile >> 7;                      // / (T/16)
  const int t0   = (tile & 127) << 4;

  const float*  xrow = x + (size_t)(b * T_ + t0 + ln) * D_;
  const __bf16* Wm   = Wt + (size_t)m * HS_ * D_;

  v8f acc[4];
  #pragma unroll
  for (int c = 0; c < 4; ++c)
    #pragma unroll
    for (int e = 0; e < 8; ++e) acc[c][e] = 0.0f;

  for (int ks = 0; ks < 24; ++ks) {
    const int kb32 = ks * 32;
    // A-frag (16x32 bf16): lanes<16 hold K {0..15}, lanes>=16 hold {16..31}
    const int d0 = kb32 + half * 16;
    V16U a;
    #pragma unroll
    for (int e = 0; e < 8; ++e) a.v[e]     = (__bf16)xrow[d0 + e];
    #pragma unroll
    for (int e = 0; e < 8; ++e) a.v[e + 8] = (__bf16)xrow[d0 + 8 + e];

    #pragma unroll
    for (int c = 0; c < 4; ++c) {
      // B-frag (32x16): element e -> row d = kb32 + e (+16 for upper half), col = ln
      const __bf16* wp = Wm + (size_t)(c * 16 + ln) * D_ + kb32 + half * 16;
      v16bf bf = *(const v16bf*)wp;
      acc[c] = __builtin_amdgcn_wmma_f32_16x16x32_bf16(
          false, a.v, false, bf, (short)0, acc[c], false, false);
    }
  }

  // Wave-uniform epilogue dispatch: one scalar branch, clean store clauses.
  if (m == 2) {
    #pragma unroll
    for (int c = 0; c < 4; ++c) {
      const int h = c * 16 + ln;
      __bf16* dcol = vT + ((size_t)b * HS_ + h) * T_ + t0 + half * 8;
      #pragma unroll
      for (int r = 0; r < 8; ++r) dcol[r] = (__bf16)acc[c][r];
    }
  } else if (m == 0) {
    #pragma unroll
    for (int c = 0; c < 4; ++c) {
      const int h = c * 16 + ln;
      __bf16* drow = qb + ((size_t)(b * T_ + t0 + half * 8)) * HS_ + h;
      #pragma unroll
      for (int r = 0; r < 8; ++r) drow[(size_t)r * HS_] = (__bf16)(acc[c][r] * 0.125f);
    }
  } else {
    #pragma unroll
    for (int c = 0; c < 4; ++c) {
      const int h = c * 16 + ln;
      __bf16* drow = kb + ((size_t)(b * T_ + t0 + half * 8)) * HS_ + h;
      #pragma unroll
      for (int r = 0; r < 8; ++r) drow[(size_t)r * HS_] = (__bf16)acc[c][r];
    }
  }
}

// ---------------------------------------------------------------------------
// Kernel 2: causal flash attention, one wave per (batch, 16-query tile).
// S^T = K @ Q^T so the exp'd scores are already in A-operand layout for P@V.
// ---------------------------------------------------------------------------
__global__ void __launch_bounds__(256) attn(const __bf16* __restrict__ qb,
                                            const __bf16* __restrict__ kb,
                                            const __bf16* __restrict__ vT,
                                            float* __restrict__ out) {
  const int lane = threadIdx.x & 31;
  const int half = lane >> 4;
  const int ln   = lane & 15;
  const int wid  = __builtin_amdgcn_readfirstlane(blockIdx.x * 8 + (threadIdx.x >> 5));
  const int b    = wid >> 7;
  const int q0   = (wid & 127) << 4;
  const float LOG2E = 1.4426950408889634f;

  // Q^T B-frags: b[e] = Q[q0+ln][chunk*32 + e (+16 upper half)] — contiguous
  const __bf16* qrow = qb + (size_t)(b * T_ + q0 + ln) * HS_;
  const v16bf qf0 = *(const v16bf*)(qrow + half * 16);
  const v16bf qf1 = *(const v16bf*)(qrow + 32 + half * 16);

  v8f o0, o1, o2, o3;
  #pragma unroll
  for (int e = 0; e < 8; ++e) { o0[e] = 0.f; o1[e] = 0.f; o2[e] = 0.f; o3[e] = 0.f; }
  float mrun = -1e30f, lrun = 0.f;

  const int nblk = (q0 >> 5) + 1;                  // 32-key blocks up to diagonal
  const __bf16* kbase = kb + (size_t)b * T_ * HS_;
  const __bf16* vbase = vT + (size_t)b * HS_ * T_;
  const int qg = q0 + ln;                          // this lane's query column

  for (int j = 0; j < nblk; ++j) {
    const int k0 = j * 32;
    if (j + 1 < nblk)
      __builtin_prefetch(kbase + (size_t)(k0 + 32 + ln) * HS_, 0, 0);

    // --- S^T tiles: s0 = keys k0..k0+15, s1 = keys k0+16..k0+31 ------------
    const __bf16* k0row = kbase + (size_t)(k0 + ln) * HS_;
    const __bf16* k1row = kbase + (size_t)(k0 + 16 + ln) * HS_;
    v8f s0, s1;
    #pragma unroll
    for (int e = 0; e < 8; ++e) { s0[e] = 0.f; s1[e] = 0.f; }

    V16U ka;
    ka.h[0] = *(const v8bf*)(k0row + half * 16);
    ka.h[1] = *(const v8bf*)(k0row + 8 + half * 16);
    s0 = __builtin_amdgcn_wmma_f32_16x16x32_bf16(false, ka.v, false, qf0, (short)0, s0, false, false);
    ka.h[0] = *(const v8bf*)(k0row + 32 + half * 16);
    ka.h[1] = *(const v8bf*)(k0row + 40 + half * 16);
    s0 = __builtin_amdgcn_wmma_f32_16x16x32_bf16(false, ka.v, false, qf1, (short)0, s0, false, false);

    ka.h[0] = *(const v8bf*)(k1row + half * 16);
    ka.h[1] = *(const v8bf*)(k1row + 8 + half * 16);
    s1 = __builtin_amdgcn_wmma_f32_16x16x32_bf16(false, ka.v, false, qf0, (short)0, s1, false, false);
    ka.h[0] = *(const v8bf*)(k1row + 32 + half * 16);
    ka.h[1] = *(const v8bf*)(k1row + 40 + half * 16);
    s1 = __builtin_amdgcn_wmma_f32_16x16x32_bf16(false, ka.v, false, qf1, (short)0, s1, false, false);

    // --- causal mask (diagonal block only; wave-uniform branch) ------------
    if (k0 + 31 > q0) {
      #pragma unroll
      for (int r = 0; r < 8; ++r) {
        const int key0 = k0 + r + half * 8;        // S^T row = key
        if (key0 > qg)      s0[r] = -1e30f;
        if (key0 + 16 > qg) s1[r] = -1e30f;
      }
    }

    // --- online softmax (per-query stats live at lane = query, both halves)
    float mloc = -1e30f;
    #pragma unroll
    for (int r = 0; r < 8; ++r) { mloc = fmaxf(mloc, s0[r]); mloc = fmaxf(mloc, s1[r]); }
    mloc = fmaxf(mloc, __shfl_xor(mloc, 16));
    const float mnew  = fmaxf(mrun, mloc);
    const float alpha = exp2f((mrun - mnew) * LOG2E);
    mrun = mnew;

    float p0[8], p1[8], ssum = 0.f;
    #pragma unroll
    for (int r = 0; r < 8; ++r) {
      p0[r] = exp2f((s0[r] - mnew) * LOG2E);
      p1[r] = exp2f((s1[r] - mnew) * LOG2E);
      ssum += p0[r] + p1[r];
    }
    ssum += __shfl_xor(ssum, 16);
    lrun = lrun * alpha + ssum;

    // rescale O: accumulator row r belongs to query r + 8*half
    #pragma unroll
    for (int r = 0; r < 8; ++r) {
      const float f = __shfl(alpha, r + half * 8, 32);
      o0[r] *= f; o1[r] *= f; o2[r] *= f; o3[r] *= f;
    }

    // --- P@V: S^T C-layout is exactly P's A-operand layout ------------------
    V16U pa;
    #pragma unroll
    for (int e = 0; e < 8; ++e) { pa.v[e] = (__bf16)p0[e]; pa.v[e + 8] = (__bf16)p1[e]; }

    const __bf16* vrow = vbase + (size_t)ln * T_ + k0 + half * 8;
    V16U vb0, vb1, vb2, vb3;
    vb0.h[0] = *(const v8bf*)(vrow);           vb0.h[1] = *(const v8bf*)(vrow + 16);
    vb1.h[0] = *(const v8bf*)(vrow + 16 * T_); vb1.h[1] = *(const v8bf*)(vrow + 16 * T_ + 16);
    vb2.h[0] = *(const v8bf*)(vrow + 32 * T_); vb2.h[1] = *(const v8bf*)(vrow + 32 * T_ + 16);
    vb3.h[0] = *(const v8bf*)(vrow + 48 * T_); vb3.h[1] = *(const v8bf*)(vrow + 48 * T_ + 16);
    o0 = __builtin_amdgcn_wmma_f32_16x16x32_bf16(false, pa.v, false, vb0.v, (short)0, o0, false, false);
    o1 = __builtin_amdgcn_wmma_f32_16x16x32_bf16(false, pa.v, false, vb1.v, (short)0, o1, false, false);
    o2 = __builtin_amdgcn_wmma_f32_16x16x32_bf16(false, pa.v, false, vb2.v, (short)0, o2, false, false);
    o3 = __builtin_amdgcn_wmma_f32_16x16x32_bf16(false, pa.v, false, vb3.v, (short)0, o3, false, false);
  }

  // --- finalize: divide by row sums, write fp32 output -----------------------
  const float invl = 1.0f / lrun;
  #pragma unroll
  for (int r = 0; r < 8; ++r) {
    const float f = __shfl(invl, r + half * 8, 32);
    const int t = q0 + r + half * 8;
    float* dst = out + (size_t)(b * T_ + t) * HS_ + ln;
    dst[0]  = o0[r] * f;
    dst[16] = o1[r] * f;
    dst[32] = o2[r] * f;
    dst[48] = o3[r] * f;
  }
}

// ---------------------------------------------------------------------------
extern "C" void kernel_launch(void* const* d_in, const int* in_sizes, int n_in,
                              void* d_out, int out_size, void* d_ws, size_t ws_size,
                              hipStream_t stream) {
  const float* x  = (const float*)d_in[0];
  const float* Wq = (const float*)d_in[1];
  const float* Wk = (const float*)d_in[2];
  const float* Wv = (const float*)d_in[3];

  char* ws = (char*)d_ws;
  __bf16* Wt = (__bf16*)(ws + WS_WT);
  __bf16* q  = (__bf16*)(ws + WS_Q);
  __bf16* k  = (__bf16*)(ws + WS_K);
  __bf16* vT = (__bf16*)(ws + WS_V);
  float* out = (float*)d_out;

  repack_w<<<(3 * HS_ * D_ + 255) / 256, 256, 0, stream>>>(Wq, Wk, Wv, Wt);
  qkv_proj<<<384, 256, 0, stream>>>(x, Wt, q, k, vT);   // 3072 waves: 1024 tiles x {Q,K,V}
  attn<<<128, 256, 0, stream>>>(q, k, vT, out);         // 1024 waves: 8 batches x 128 q-tiles
}
